// SchNetModel_72980084294216
// MI455X (gfx1250) — matrix operand order repaired and checked by reading, not compiled, not run on hardware
//
#include <hip/hip_runtime.h>
#include <hip/hip_bf16.h>

typedef __attribute__((ext_vector_type(16))) _Float16 v16h;
typedef __attribute__((ext_vector_type(8)))  float    v8f;

#define HID 64          // HIDDEN == FILTERS == 64
#define NB  16          // gaussian bases
#define LDP 72          // padded LDS row stride (floats); col 64 holds envelope
#define CUTOFF 5.0f
#define GSTEP  (CUTOFF / (NB - 1))          // 1/3
#define GCOEFF (-0.5f / (GSTEP * GSTEP))    // -4.5
#define PI_OVER_CUT (3.14159265358979f / CUTOFF)
#define LN2 0.6931471805599453f

// ---------------------------------------------------------------- helpers

__device__ __forceinline__ float ssp(float x) {
    float ax = fabsf(x);
    return __logf(1.0f + __expf(-ax)) + fmaxf(x, 0.0f) - LN2;
}

// B-fragment for D = A @ Wt^T : B[k][n] = Wt[n][k].  Wt row-major [64 x K].
__device__ __forceinline__ v16h load_bfrag(const float* Wt, int n0, int kb,
                                           int K, int Kvalid) {
    int lane  = threadIdx.x & 31;
    int n     = n0 + (lane & 15);
    int kbase = kb + ((lane >> 4) << 3);
    v16h b;
#pragma unroll
    for (int j = 0; j < 16; ++j) {
        int k = kbase + j + ((j >= 8) ? 8 : 0);
        float v = (k < Kvalid) ? Wt[n * K + k] : 0.0f;
        b[j] = (_Float16)v;
    }
    return b;
}

// A-fragment 16x32 f16 from row-major f32, rows row0..row0+15 (clamped).
__device__ __forceinline__ v16h load_afrag_g(const float* X, int row0,
                                             int maxrow, int kb) {
    int lane = threadIdx.x & 31;
    int r = row0 + (lane & 15);
    if (r > maxrow) r = maxrow;
    int kbase = kb + ((lane >> 4) << 3);
    const float* p = X + r * HID;
    v16h a;
#pragma unroll
    for (int j = 0; j < 16; ++j) {
        int k = kbase + j + ((j >= 8) ? 8 : 0);
        a[j] = (_Float16)p[k];
    }
    return a;
}

// Same A-fragment but sourced from wave-private LDS f32 [16][LDP].
__device__ __forceinline__ v16h load_afrag_lds(const float* wl, int kb) {
    int lane = threadIdx.x & 31;
    int m = lane & 15;
    int kbase = kb + ((lane >> 4) << 3);
    v16h a;
#pragma unroll
    for (int j = 0; j < 16; ++j) {
        int k = kbase + j + ((j >= 8) ? 8 : 0);
        a[j] = (_Float16)wl[m * LDP + k];
    }
    return a;
}

#define WMMA_F16(A, B, C)                                                     \
    __builtin_amdgcn_wmma_f32_16x16x32_f16(false, (A), false, (B), (short)0,  \
                                           (C), false, false)

// ---------------------------------------------------------------- kernels

__global__ void embed_kernel(const float* __restrict__ attrs,
                             const float* __restrict__ Wv,
                             float* __restrict__ h, int N, int S) {
    int i = blockIdx.x * blockDim.x + threadIdx.x;
    if (i >= N * HID) return;
    int n = i >> 6;
    int j = i & 63;
    float acc = 0.0f;
    for (int s = 0; s < S; ++s) acc += attrs[n * S + s] * Wv[j * S + s];
    h[i] = acc;
}

__global__ void edge_d_kernel(const float* __restrict__ pos,
                              const float* __restrict__ shifts,
                              const int* __restrict__ src,
                              const int* __restrict__ dst,
                              float* __restrict__ d_arr, int E) {
    int e = blockIdx.x * blockDim.x + threadIdx.x;
    if (e >= E) return;
    int s = src[e], t = dst[e];
    float dx = pos[t * 3 + 0] - pos[s * 3 + 0] + shifts[e * 3 + 0];
    float dy = pos[t * 3 + 1] - pos[s * 3 + 1] + shifts[e * 3 + 1];
    float dz = pos[t * 3 + 2] - pos[s * 3 + 2] + shifts[e * 3 + 2];
    d_arr[e] = sqrtf(dx * dx + dy * dy + dz * dz);
}

__global__ void zero_kernel(float* __restrict__ p, int n) {
    int i = blockIdx.x * blockDim.x + threadIdx.x;
    if (i < n) p[i] = 0.0f;
}

// xf = h @ lin1^T   ([N,64]@[64,64], WMMA, one 16-row tile per wave)
__global__ void xf_gemm_kernel(const float* __restrict__ h,
                               const float* __restrict__ lin1,
                               float* __restrict__ xf, int N) {
    int lane = threadIdx.x & 31;
    int wid = (blockIdx.x * blockDim.x + threadIdx.x) >> 5;
    int nwaves = (gridDim.x * blockDim.x) >> 5;
    int ntiles = (N + 15) >> 4;

    v16h B[4][2];
#pragma unroll
    for (int c = 0; c < 4; ++c)
#pragma unroll
        for (int kb = 0; kb < 2; ++kb)
            B[c][kb] = load_bfrag(lin1, 16 * c, kb * 32, HID, HID);

    for (int t = wid; t < ntiles; t += nwaves) {
        int row0 = t * 16;
        v8f acc[4] = {};
#pragma unroll
        for (int kb = 0; kb < 2; ++kb) {
            v16h a = load_afrag_g(h, row0, N - 1, kb * 32);
#pragma unroll
            for (int c = 0; c < 4; ++c) acc[c] = WMMA_F16(a, B[c][kb], acc[c]);
        }
        int rbase = row0 + ((lane >> 4) << 3);
        int col = (lane & 15);
        if (row0 + 16 <= N) {            // wave-uniform: branch-free stores
#pragma unroll
            for (int c = 0; c < 4; ++c)
#pragma unroll
                for (int r = 0; r < 8; ++r)
                    xf[(rbase + r) * HID + 16 * c + col] = acc[c][r];
        } else {
#pragma unroll
            for (int c = 0; c < 4; ++c)
#pragma unroll
                for (int r = 0; r < 8; ++r)
                    if (rbase + r < N)
                        xf[(rbase + r) * HID + 16 * c + col] = acc[c][r];
        }
    }
}

// Fused per-layer edge kernel: gaussian basis -> filter MLP (WMMA x2 with
// ssp in between) -> cosine cutoff -> wave-private LDS -> coalesced gather
// of xf[src] -> atomic scatter-add into agg[dst].
__global__ void __launch_bounds__(256)
edge_fused_kernel(const float* __restrict__ d_arr,
                  const int* __restrict__ src, const int* __restrict__ dst,
                  const float* __restrict__ xf,
                  const float* __restrict__ w1, const float* __restrict__ b1,
                  const float* __restrict__ w2, const float* __restrict__ b2,
                  float* __restrict__ agg, int E) {
    __shared__ float lds[8][16 * LDP];
    int lane = threadIdx.x & 31;
    float* wl = &lds[threadIdx.x >> 5][0];
    int wid = (blockIdx.x * blockDim.x + threadIdx.x) >> 5;
    int nwaves = (gridDim.x * blockDim.x) >> 5;
    int ntiles = (E + 15) >> 4;

    v16h B1[4];
#pragma unroll
    for (int c = 0; c < 4; ++c) B1[c] = load_bfrag(w1, 16 * c, 0, NB, NB);
    v16h B2[4][2];
#pragma unroll
    for (int c = 0; c < 4; ++c)
#pragma unroll
        for (int kb = 0; kb < 2; ++kb)
            B2[c][kb] = load_bfrag(w2, 16 * c, kb * 32, HID, HID);

    const int l15 = lane & 15;
    const int hi8 = (lane >> 4) << 3;

    for (int t = wid; t < ntiles; t += nwaves) {
        int e0 = t * 16;
        bool full = (e0 + 16 <= E);
        // distance for row m = lane&15 (lanes m and m+16 hold same row)
        int em = e0 + l15; if (em >= E) em = E - 1;
        float dm = d_arr[em];
        // envelope -> spare LDS column (written redundantly by both halves)
        wl[l15 * LDP + 64] = 0.5f * (__cosf(dm * PI_OVER_CUT) + 1.0f);
        // src/dst indices: lanes 0-15 = src, lanes 16-31 = dst
        const int* sd = (lane < 16) ? src : dst;
        int idxv = sd[em];
        // A = gaussian features, zero-padded K 16..31
        v16h a;
#pragma unroll
        for (int j = 0; j < 16; ++j) {
            int k = hi8 + j + ((j >= 8) ? 8 : 0);
            float v = 0.0f;
            if (k < NB) { float u = dm - (float)k * GSTEP; v = __expf(GCOEFF * u * u); }
            a[j] = (_Float16)v;
        }
        // GEMM1 + bias + ssp -> LDS (f32, [edge][feat])
        v8f acc[4] = {};
#pragma unroll
        for (int c = 0; c < 4; ++c) acc[c] = WMMA_F16(a, B1[c], acc[c]);
#pragma unroll
        for (int c = 0; c < 4; ++c) {
            float bn = b1[16 * c + l15];
#pragma unroll
            for (int r = 0; r < 8; ++r)
                wl[(r + hi8) * LDP + 16 * c + l15] = ssp(acc[c][r] + bn);
        }
        // GEMM2 (transpose through LDS) + bias, scale by envelope
        v8f accW[4] = {};
#pragma unroll
        for (int kb = 0; kb < 2; ++kb) {
            v16h a2 = load_afrag_lds(wl, kb * 32);
#pragma unroll
            for (int c = 0; c < 4; ++c) accW[c] = WMMA_F16(a2, B2[c][kb], accW[c]);
        }
        float cenv[8];
#pragma unroll
        for (int r = 0; r < 8; ++r) cenv[r] = wl[(r + hi8) * LDP + 64];
#pragma unroll
        for (int c = 0; c < 4; ++c) {
            float bn = b2[16 * c + l15];
#pragma unroll
            for (int r = 0; r < 8; ++r)
                wl[(r + hi8) * LDP + 16 * c + l15] = (accW[c][r] + bn) * cenv[r];
        }
        // gather * filter -> atomic scatter; indices pulled to SGPRs
        if (full) {
#pragma unroll 4
            for (int e = 0; e < 16; ++e) {
                int s  = __builtin_amdgcn_readlane(idxv, e);
                int tt = __builtin_amdgcn_readlane(idxv, e + 16);
                float f0 = wl[e * LDP + lane];
                float f1 = wl[e * LDP + lane + 32];
                float x0 = xf[s * HID + lane];
                float x1 = xf[s * HID + lane + 32];
                atomicAdd(&agg[tt * HID + lane],      x0 * f0);
                atomicAdd(&agg[tt * HID + lane + 32], x1 * f1);
            }
        } else {
            for (int e = 0; e < 16; ++e) {
                if (e0 + e >= E) break;          // uniform across wave
                int s  = __builtin_amdgcn_readlane(idxv, e);
                int tt = __builtin_amdgcn_readlane(idxv, e + 16);
                float f0 = wl[e * LDP + lane];
                float f1 = wl[e * LDP + lane + 32];
                float x0 = xf[s * HID + lane];
                float x1 = xf[s * HID + lane + 32];
                atomicAdd(&agg[tt * HID + lane],      x0 * f0);
                atomicAdd(&agg[tt * HID + lane + 32], x1 * f1);
            }
        }
    }
}

// m = ssp(agg@lin2^T + b2) @ lin^T + b ; h += m   (two chained WMMA GEMMs)
__global__ void __launch_bounds__(256)
node_update_kernel(const float* __restrict__ agg,
                   const float* __restrict__ lin2, const float* __restrict__ b2,
                   const float* __restrict__ linw, const float* __restrict__ bw,
                   float* __restrict__ h, int N) {
    __shared__ float lds[8][16 * LDP];
    int lane = threadIdx.x & 31;
    float* wl = &lds[threadIdx.x >> 5][0];
    int wid = (blockIdx.x * blockDim.x + threadIdx.x) >> 5;
    int nwaves = (gridDim.x * blockDim.x) >> 5;
    int ntiles = (N + 15) >> 4;

    v16h BL2[4][2], BLW[4][2];
#pragma unroll
    for (int c = 0; c < 4; ++c)
#pragma unroll
        for (int kb = 0; kb < 2; ++kb) {
            BL2[c][kb] = load_bfrag(lin2, 16 * c, kb * 32, HID, HID);
            BLW[c][kb] = load_bfrag(linw, 16 * c, kb * 32, HID, HID);
        }

    const int l15 = lane & 15;
    const int hi8 = (lane >> 4) << 3;

    for (int t = wid; t < ntiles; t += nwaves) {
        int row0 = t * 16;
        v8f acc[4] = {};
#pragma unroll
        for (int kb = 0; kb < 2; ++kb) {
            v16h a = load_afrag_g(agg, row0, N - 1, kb * 32);
#pragma unroll
            for (int c = 0; c < 4; ++c) acc[c] = WMMA_F16(a, BL2[c][kb], acc[c]);
        }
#pragma unroll
        for (int c = 0; c < 4; ++c) {
            float bn = b2[16 * c + l15];
#pragma unroll
            for (int r = 0; r < 8; ++r)
                wl[(r + hi8) * LDP + 16 * c + l15] = ssp(acc[c][r] + bn);
        }
        v8f acc2[4] = {};
#pragma unroll
        for (int kb = 0; kb < 2; ++kb) {
            v16h a2 = load_afrag_lds(wl, kb * 32);
#pragma unroll
            for (int c = 0; c < 4; ++c) acc2[c] = WMMA_F16(a2, BLW[c][kb], acc2[c]);
        }
        int rbase = row0 + hi8;
        if (row0 + 16 <= N) {            // wave-uniform: branch-free path
#pragma unroll
            for (int c = 0; c < 4; ++c) {
                float bn = bw[16 * c + l15];
#pragma unroll
                for (int r = 0; r < 8; ++r)
                    h[(rbase + r) * HID + 16 * c + l15] += acc2[c][r] + bn;
            }
        } else {
#pragma unroll
            for (int c = 0; c < 4; ++c) {
                float bn = bw[16 * c + l15];
#pragma unroll
                for (int r = 0; r < 8; ++r)
                    if (rbase + r < N)
                        h[(rbase + r) * HID + 16 * c + l15] += acc2[c][r] + bn;
            }
        }
    }
}

// readout MLP + scatter-mean accumulation
__global__ void readout_kernel(const float* __restrict__ h,
                               const float* __restrict__ ow1,
                               const float* __restrict__ ob1,
                               const float* __restrict__ ow2,
                               const float* __restrict__ ob2,
                               const int* __restrict__ batch,
                               float* __restrict__ sums,
                               float* __restrict__ counts, int N) {
    int n = blockIdx.x * blockDim.x + threadIdx.x;
    if (n >= N) return;
    float o = ob2[0];
    for (int j = 0; j < HID / 2; ++j) {
        float s = ob1[j];
        for (int k = 0; k < HID; ++k) s += h[n * HID + k] * ow1[j * HID + k];
        o += ssp(s) * ow2[j];
    }
    int g = batch[n];
    atomicAdd(&sums[g], o);
    atomicAdd(&counts[g], 1.0f);
}

__global__ void finalize_kernel(const float* __restrict__ sums,
                                const float* __restrict__ counts,
                                float* __restrict__ out, int G) {
    int g = blockIdx.x * blockDim.x + threadIdx.x;
    if (g < G) out[g] = sums[g] / fmaxf(counts[g], 1.0f);
}

// ---------------------------------------------------------------- launch

extern "C" void kernel_launch(void* const* d_in, const int* in_sizes, int n_in,
                              void* d_out, int out_size, void* d_ws, size_t ws_size,
                              hipStream_t stream) {
    const float* positions  = (const float*)d_in[0];
    const float* shifts     = (const float*)d_in[1];
    const float* node_attrs = (const float*)d_in[2];
    const int*   batch      = (const int*)d_in[3];
    const int*   edge_index = (const int*)d_in[4];
    const float* W_v_w      = (const float*)d_in[5];
    const float* mlp_w1     = (const float*)d_in[6];
    const float* mlp_b1     = (const float*)d_in[7];
    const float* mlp_w2     = (const float*)d_in[8];
    const float* mlp_b2     = (const float*)d_in[9];
    const float* lin1_w     = (const float*)d_in[10];
    const float* lin2_w     = (const float*)d_in[11];
    const float* lin2_b     = (const float*)d_in[12];
    const float* lin_w      = (const float*)d_in[13];
    const float* lin_b      = (const float*)d_in[14];
    const float* out_w1     = (const float*)d_in[15];
    const float* out_b1     = (const float*)d_in[16];
    const float* out_w2     = (const float*)d_in[17];
    const float* out_b2     = (const float*)d_in[18];
    float* out = (float*)d_out;

    const int N = in_sizes[0] / 3;
    const int E = in_sizes[4] / 2;
    const int S = in_sizes[2] / N;
    const int G = out_size;   // N_OUT == 1

    const int* src = edge_index;
    const int* dst = edge_index + E;

    float* h      = (float*)d_ws;
    float* xf     = h     + (size_t)N * HID;
    float* agg    = xf    + (size_t)N * HID;
    float* d_arr  = agg   + (size_t)N * HID;
    float* sums   = d_arr + E;
    float* counts = sums + G;

    const int ntilesE = (E + 15) >> 4;
    const int ntilesN = (N + 15) >> 4;
    int blkE = (ntilesE + 7) / 8; if (blkE > 4096) blkE = 4096;
    int blkN = (ntilesN + 7) / 8; if (blkN > 2048) blkN = 2048;

    embed_kernel<<<(N * HID + 255) / 256, 256, 0, stream>>>(
        node_attrs, W_v_w, h, N, S);
    edge_d_kernel<<<(E + 255) / 256, 256, 0, stream>>>(
        positions, shifts, src, dst, d_arr, E);

    for (int l = 0; l < 2; ++l) {
        xf_gemm_kernel<<<blkN, 256, 0, stream>>>(h, lin1_w + l * HID * HID, xf, N);
        zero_kernel<<<(N * HID + 255) / 256, 256, 0, stream>>>(agg, N * HID);
        edge_fused_kernel<<<blkE, 256, 0, stream>>>(
            d_arr, src, dst, xf,
            mlp_w1 + l * HID * NB, mlp_b1 + l * HID,
            mlp_w2 + l * HID * HID, mlp_b2 + l * HID, agg, E);
        node_update_kernel<<<blkN, 256, 0, stream>>>(
            agg, lin2_w + l * HID * HID, lin2_b + l * HID,
            lin_w + l * HID * HID, lin_b + l * HID, h, N);
    }

    zero_kernel<<<(2 * G + 255) / 256, 256, 0, stream>>>(sums, 2 * G);
    readout_kernel<<<(N + 255) / 256, 256, 0, stream>>>(
        h, out_w1, out_b1, out_w2, out_b2, batch, sums, counts, N);
    finalize_kernel<<<(G + 255) / 256, 256, 0, stream>>>(sums, counts, out, G);
}
